// GraphAttn_28157805593322
// MI455X (gfx1250) — matrix operand order repaired and checked by reading, not compiled, not run on hardware
//
#include <hip/hip_runtime.h>
#include <hip/hip_bf16.h>

// ---------------------------------------------------------------------------
// Problem constants (from reference): B=16, N=2048 (nodes), K=2048, F=64 feat
// ---------------------------------------------------------------------------
#define BB   16
#define NN   2048
#define KK   2048
#define FF   64
#define NEG_SLOPE 0.2f

typedef __attribute__((ext_vector_type(16))) __bf16        v16bf;
typedef __attribute__((ext_vector_type(8)))  float         v8f;
typedef __attribute__((ext_vector_type(4)))  unsigned int  v4u;

struct Op32B { v4u lo, hi; };   // 32 bytes == one bf16 WMMA operand per lane

__device__ __forceinline__ unsigned short f2bf(float f) {
    unsigned u = __builtin_bit_cast(unsigned, f);
    u += 0x7FFFu + ((u >> 16) & 1u);          // round-to-nearest-even
    return (unsigned short)(u >> 16);
}
__device__ __forceinline__ unsigned pk2(float a, float b) {
    return (unsigned)f2bf(a) | ((unsigned)f2bf(b) << 16);
}
__device__ __forceinline__ v4u pk8(float4 x, float4 y) {
    v4u r;
    r[0] = pk2(x.x, x.y); r[1] = pk2(x.z, x.w);
    r[2] = pk2(y.x, y.y); r[3] = pk2(y.z, y.w);
    return r;
}
// Load one 32-byte WMMA operand (two ds_load_b128) from packed LDS.
__device__ __forceinline__ v16bf ld_op(const v4u* base, int idx) {
    Op32B p{base[idx], base[idx + 1]};
    return __builtin_bit_cast(v16bf, p);
}

// Packed A-operand LDS layout for a 128(M) x 32(K) bf16 tile:
//   [strip = m>>4][lane = (m&15) + 16*grp][half][8 elems], grp/half from K:
//   K segment s (= k>>3, 0..3) -> grp = s&1, half = s>>1  (matches ISA 16-bit
//   A layout: elem e of lane(grp) holds K = (e>>3)*16 + grp*8 + (e&7)).
__device__ __forceinline__ int a_slot(int row, int seg) {
    return ((row >> 4) * 32 + ((row & 15) + 16 * (seg & 1))) * 2 + (seg >> 1);
}

// ---------------------------------------------------------------------------
// Kernel 1: Y[b] = W @ X[b] + bias      (M=2048, K=2048, N=64; 16 batches)
// block = 256 thr (8 waves). Block tile: 128(M) x 64(N), K chunk 32.
// ---------------------------------------------------------------------------
__global__ __launch_bounds__(256) void k_transform(
    const float* __restrict__ X, const float* __restrict__ W,
    const float* __restrict__ bias, float* __restrict__ Y) {
    __shared__ v4u As4[8 * 32 * 2];   // packed A operands (8 KB)
    __shared__ v4u Bt4[64 * 4];       // B transposed [n][k], 32 ushort/col (4 KB)

    const int b    = blockIdx.y;
    const int m0   = blockIdx.x * 128;
    const int tid  = threadIdx.x;
    const int lane = tid & 31;
    const int wave = tid >> 5;
    const int grp  = lane >> 4;
    const int l16  = lane & 15;
    unsigned* Btu  = (unsigned*)Bt4;

    v8f acc[4] = {};

    for (int k0 = 0; k0 < KK; k0 += 32) {
        // stage W: 512 segments of 8 consecutive K -> packed A layout
#pragma unroll
        for (int it = 0; it < 2; ++it) {
            int p   = tid + it * 256;
            int row = p >> 2, seg = p & 3;
            const float4* src =
                (const float4*)(W + (size_t)(m0 + row) * KK + k0 + seg * 8);
            As4[a_slot(row, seg)] = pk8(src[0], src[1]);
        }
        // stage X transposed: Bt[f][k], packed bf16 pairs as u32
#pragma unroll
        for (int it = 0; it < 4; ++it) {
            int p  = tid + it * 256;
            int f  = p & 63, kp = p >> 6;            // kp = 0..15, k = 2*kp
            size_t base = ((size_t)b * KK + (k0 + 2 * kp)) * FF + f;
            Btu[f * 16 + ((kp & 3) | ((kp & 4) << 1) | ((kp & 8) >> 1))] = pk2(X[base], X[base + FF]);
        }
        if (k0 + 32 < KK)   // global_prefetch_b8 of next W K-slice
            __builtin_prefetch(W + (size_t)(m0 + (tid & 127)) * KK + (k0 + 32), 0, 1);
        __syncthreads();

        v16bf a = ld_op(As4, (wave * 32 + lane) * 2);
#pragma unroll
        for (int nt = 0; nt < 4; ++nt) {
            v16bf bb = ld_op(Bt4, (nt * 16 + l16) * 4 + grp * 2);
            acc[nt] = __builtin_amdgcn_wmma_f32_16x16x32_bf16(
                false, a, false, bb, (short)0, acc[nt], false, false);
        }
        __syncthreads();
    }

#pragma unroll
    for (int r = 0; r < 8; ++r) {
        int m = m0 + wave * 16 + r + 8 * grp;
        float bv = bias[m];
#pragma unroll
        for (int nt = 0; nt < 4; ++nt)
            Y[((size_t)b * NN + m) * FF + nt * 16 + l16] = acc[nt][r] + bv;
    }
}

// ---------------------------------------------------------------------------
// Kernel 2: rowdot  out[row] = sum_f w[f] * src[row*64+f]   (rows = B*N)
// ---------------------------------------------------------------------------
__global__ __launch_bounds__(64) void k_rowdot(
    const float* __restrict__ src, const float* __restrict__ wvec,
    float* __restrict__ out) {
    __shared__ float red[64];
    const int row = blockIdx.x;
    const int f   = threadIdx.x;
    red[f] = src[(size_t)row * FF + f] * wvec[f];
    __syncthreads();
    for (int s = 32; s > 0; s >>= 1) {
        if (f < s) red[f] += red[f + s];
        __syncthreads();
    }
    if (f == 0) out[row] = red[0];
}

// ---------------------------------------------------------------------------
// Kernel 3: s2[b,m] = (W @ vn[b])[m] + conv_b[m]*sum(att2)
// batched GEMV as GEMM: M=2048, K=2048, N=16 (batches). One WMMA N-tile.
// ---------------------------------------------------------------------------
__global__ __launch_bounds__(256) void k_s2(
    const float* __restrict__ W, const float* __restrict__ vn,
    const float* __restrict__ bias, const float* __restrict__ att2,
    float* __restrict__ s2) {
    __shared__ v4u As4[8 * 32 * 2];
    __shared__ v4u Bt4[16 * 4];       // [b][k] transposed

    const int m0   = blockIdx.x * 128;
    const int tid  = threadIdx.x;
    const int lane = tid & 31;
    const int wave = tid >> 5;
    const int grp  = lane >> 4;
    const int l16  = lane & 15;
    unsigned* Btu  = (unsigned*)Bt4;

    v8f acc = {};

    for (int k0 = 0; k0 < KK; k0 += 32) {
#pragma unroll
        for (int it = 0; it < 2; ++it) {
            int p   = tid + it * 256;
            int row = p >> 2, seg = p & 3;
            const float4* src =
                (const float4*)(W + (size_t)(m0 + row) * KK + k0 + seg * 8);
            As4[a_slot(row, seg)] = pk8(src[0], src[1]);
        }
        if (tid < 256) {                 // 256 u32 pairs: 16 batches x 16 kp
            int kp = tid & 15, c = tid >> 4;
            size_t base = (size_t)c * NN + k0 + 2 * kp;
            Btu[c * 16 + ((kp & 3) | ((kp & 4) << 1) | ((kp & 8) >> 1))] = pk2(vn[base], vn[base + 1]);
        }
        __syncthreads();

        v16bf a  = ld_op(As4, (wave * 32 + lane) * 2);
        v16bf bb = ld_op(Bt4, l16 * 4 + grp * 2);
        acc = __builtin_amdgcn_wmma_f32_16x16x32_bf16(
            false, a, false, bb, (short)0, acc, false, false);
        __syncthreads();
    }

    float sa = 0.f;
    for (int f = 0; f < FF; ++f) sa += att2[f];

#pragma unroll
    for (int r = 0; r < 8; ++r) {
        int m = m0 + wave * 16 + r + 8 * grp;
        s2[(size_t)l16 * NN + m] = acc[r] + bias[m] * sa;   // [b][m]
    }
}

// ---------------------------------------------------------------------------
// Kernel 4: per-column softmax stats over i (softmax axis=1 of the reference)
// ---------------------------------------------------------------------------
__global__ __launch_bounds__(256) void k_softmax_stats(
    const float* __restrict__ s1, const float* __restrict__ s2,
    float* __restrict__ cmax, float* __restrict__ crdn) {
    __shared__ float red[256];
    const int b   = blockIdx.x >> 11;
    const int j   = blockIdx.x & (NN - 1);
    const int tid = threadIdx.x;
    const float s1v = s1[(size_t)b * NN + j];

    float lmax = -3.402823466e+38f;
    for (int i = tid; i < NN; i += 256) {
        float v = s1v + s2[(size_t)b * NN + i];
        v = (v > 0.f) ? v : NEG_SLOPE * v;
        lmax = fmaxf(lmax, v);
    }
    red[tid] = lmax;
    __syncthreads();
    for (int s = 128; s > 0; s >>= 1) {
        if (tid < s) red[tid] = fmaxf(red[tid], red[tid + s]);
        __syncthreads();
    }
    const float mx = red[0];
    __syncthreads();

    float lsum = 0.f;
    for (int i = tid; i < NN; i += 256) {
        float v = s1v + s2[(size_t)b * NN + i];
        v = (v > 0.f) ? v : NEG_SLOPE * v;
        lsum += __expf(v - mx);
    }
    red[tid] = lsum;
    __syncthreads();
    for (int s = 128; s > 0; s >>= 1) {
        if (tid < s) red[tid] += red[tid + s];
        __syncthreads();
    }
    if (tid == 0) {
        cmax[(size_t)b * NN + j] = mx;
        crdn[(size_t)b * NN + j] = 1.0f / red[0];
    }
}

// ---------------------------------------------------------------------------
// Kernel 5: out[b,i,f] = sum_j P[b,i,j] * Y[b,j,f]
//   P regenerated on the fly (v_exp_f32) straight into packed A layout.
// ---------------------------------------------------------------------------
__global__ __launch_bounds__(256) void k_attend(
    const float* __restrict__ s1, const float* __restrict__ s2,
    const float* __restrict__ cmax, const float* __restrict__ crdn,
    const float* __restrict__ Y, float* __restrict__ out) {
    __shared__ v4u Ps4[8 * 32 * 2];   // packed P (A operand) tile
    __shared__ v4u Yt4[64 * 4];       // Y transposed [f][j]
    __shared__ float S2s[128];
    __shared__ float Cs1[32], Cmx[32], Crd[32];

    const int b    = blockIdx.y;
    const int i0   = blockIdx.x * 128;
    const int tid  = threadIdx.x;
    const int lane = tid & 31;
    const int wave = tid >> 5;
    const int grp  = lane >> 4;
    const int l16  = lane & 15;
    unsigned* Ytu  = (unsigned*)Yt4;

    if (tid < 128) S2s[tid] = s2[(size_t)b * NN + i0 + tid];

    v8f acc[4] = {};

    for (int j0 = 0; j0 < NN; j0 += 32) {
        if (tid < 32) {
            Cs1[tid] = s1[(size_t)b * NN + j0 + tid];
            Cmx[tid] = cmax[(size_t)b * NN + j0 + tid];
            Crd[tid] = crdn[(size_t)b * NN + j0 + tid];
        }
        __syncthreads();

        // generate P tile by 8-wide K segments, directly in packed layout
#pragma unroll
        for (int it = 0; it < 2; ++it) {
            int p   = tid + it * 256;
            int row = p >> 2, seg = p & 3;
            float s2v = S2s[row];
            float pv[8];
#pragma unroll
            for (int q = 0; q < 8; ++q) {
                int jj = seg * 8 + q;
                float v = Cs1[jj] + s2v;
                v = (v > 0.f) ? v : NEG_SLOPE * v;
                pv[q] = __expf(v - Cmx[jj]) * Crd[jj];
            }
            v4u r;
            r[0] = pk2(pv[0], pv[1]); r[1] = pk2(pv[2], pv[3]);
            r[2] = pk2(pv[4], pv[5]); r[3] = pk2(pv[6], pv[7]);
            Ps4[a_slot(row, seg)] = r;
        }
        // stage Y transposed: Yt[f][j]
#pragma unroll
        for (int it = 0; it < 4; ++it) {
            int p  = tid + it * 256;
            int f  = p & 63, kp = p >> 6;
            size_t base = ((size_t)b * NN + (j0 + 2 * kp)) * FF + f;
            Ytu[f * 16 + ((kp & 3) | ((kp & 4) << 1) | ((kp & 8) >> 1))] = pk2(Y[base], Y[base + FF]);
        }
        __syncthreads();

        v16bf a = ld_op(Ps4, (wave * 32 + lane) * 2);
#pragma unroll
        for (int nt = 0; nt < 4; ++nt) {
            v16bf bb = ld_op(Yt4, (nt * 16 + l16) * 4 + grp * 2);
            acc[nt] = __builtin_amdgcn_wmma_f32_16x16x32_bf16(
                false, a, false, bb, (short)0, acc[nt], false, false);
        }
        __syncthreads();
    }

#pragma unroll
    for (int r = 0; r < 8; ++r) {
        int i = i0 + wave * 16 + r + 8 * grp;
#pragma unroll
        for (int nt = 0; nt < 4; ++nt)
            out[((size_t)b * NN + i) * FF + nt * 16 + l16] = acc[nt][r];
    }
}

// ---------------------------------------------------------------------------
extern "C" void kernel_launch(void* const* d_in, const int* in_sizes, int n_in,
                              void* d_out, int out_size, void* d_ws, size_t ws_size,
                              hipStream_t stream) {
    (void)in_sizes; (void)n_in; (void)out_size; (void)ws_size;
    const float* x    = (const float*)d_in[0];   // (16,2048,4,16)
    const float* n    = (const float*)d_in[1];   // (16,2048,4,16)
    const float* W    = (const float*)d_in[2];   // (2048,2048,1,1)
    const float* bias = (const float*)d_in[3];   // (2048,)
    const float* att1 = (const float*)d_in[4];   // (1,64)
    const float* att2 = (const float*)d_in[5];   // (1,64)
    float*       out  = (float*)d_out;           // (16,2048,4,16)

    float* Y    = (float*)d_ws;                  // B*N*F
    float* s1   = Y    + (size_t)BB * NN * FF;
    float* s2   = s1   + (size_t)BB * NN;
    float* vn   = s2   + (size_t)BB * NN;
    float* cmax = vn   + (size_t)BB * NN;
    float* crdn = cmax + (size_t)BB * NN;

    k_transform<<<dim3(NN / 128, BB), 256, 0, stream>>>(x, W, bias, Y);
    k_rowdot<<<BB * NN, 64, 0, stream>>>(n, att2, vn);
    k_rowdot<<<BB * NN, 64, 0, stream>>>(Y, att1, s1);
    k_s2<<<NN / 128, 256, 0, stream>>>(W, vn, bias, att2, s2);
    k_softmax_stats<<<BB * NN, 256, 0, stream>>>(s1, s2, cmax, crdn);
    k_attend<<<dim3(NN / 128, BB), 256, 0, stream>>>(s1, s2, cmax, crdn, Y, out);
}
